// GNNEncoder_5488968204772
// MI455X (gfx1250) — matrix operand order repaired and checked by reading, not compiled, not run on hardware
//
#include <hip/hip_runtime.h>
#include <cstdint>
#include <cstddef>

// ---------------------------------------------------------------------------
// CDNA5 (gfx1250) heterogeneous-GNN encoder.
//   * Dense x@W via v_wmma_f32_16x16x32_bf16. Tiles are staged into LDS in
//     WMMA *fragment layout*, so each lane loads its whole v16bf fragment as
//     two contiguous ds_load_b128 (no per-element gathers / dscnt chains).
//   * TK=64: two WMMAs per barrier interval (K is always a multiple of 64).
//   * GAT fusion: a_d = x @ (Wd @ ad) -- the hd GEMM is never materialized.
//   * Edge scatters: one wave32 per edge, lane-strided features, coalesced
//     global_atomic_add_f32; float segment-max via int/uint atomic punning.
// Roofline: ~50 GFLOP of GEMM (trivial for WMMA) vs ~5 GB/layer of irregular
// scatter traffic -> the scatter phase is the 23.3 TB/s HBM-bound bottleneck.
// ---------------------------------------------------------------------------

typedef __attribute__((ext_vector_type(16))) __bf16 v16bf;
typedef __attribute__((ext_vector_type(8)))  __bf16 v8bf;
typedef __attribute__((ext_vector_type(4)))  __bf16 v4bf;
typedef __attribute__((ext_vector_type(8)))  float  v8f;

static inline int cdiv_h(int a, int b) { return (a + b - 1) / b; }

// ---------------------------- utility kernels ------------------------------

__global__ void fill_kernel(float* __restrict__ p, float v, size_t n) {
  size_t i = (size_t)blockIdx.x * blockDim.x + threadIdx.x;
  size_t stride = (size_t)gridDim.x * blockDim.x;
  for (; i < n; i += stride) p[i] = v;
}

__global__ void relu_kernel(float* __restrict__ p, size_t n) {
  size_t i = (size_t)blockIdx.x * blockDim.x + threadIdx.x;
  size_t stride = (size_t)gridDim.x * blockDim.x;
  for (; i < n; i += stride) p[i] = fmaxf(p[i], 0.f);
}

__global__ void bias_add_kernel(float* __restrict__ G, const float* __restrict__ b,
                                int N, int D) {
  size_t n = (size_t)N * D;
  size_t i = (size_t)blockIdx.x * blockDim.x + threadIdx.x;
  size_t stride = (size_t)gridDim.x * blockDim.x;
  for (; i < n; i += stride) G[i] += b[i & (D - 1)];   // D is a power of two
}

// float atomic max via signed/unsigned integer atomics (standard punning)
__device__ __forceinline__ float atomicMaxF(float* addr, float val) {
  if (val >= 0.f)
    return __int_as_float(atomicMax((int*)addr, __float_as_int(val)));
  return __uint_as_float(atomicMin((unsigned int*)addr, __float_as_uint(val)));
}

// ------------------------------ WMMA GEMM ----------------------------------
// C[M,N] (+=) A[M,K] @ W[K,N] (+ bias).  Requires K % 64 == 0, N % 32 == 0
// (true for all call sites: K in {512,256}, N in {256,128}).
// Block = 256 threads = 8 waves; block tile 64(M) x 32(N), TK = 64.
// Wave (w&3) -> 16-row M block, (w>>2) -> 16-col N block; each wave owns one
// 16x16 fp32 accumulator and issues 2 WMMAs per K-tile.
//
// LDS is written in WMMA fragment layout:
//   AsF[kslab(2)][mblk(4)][lane(32)][e(16)]  (8 KB)
//   BsF[kslab(2)][nblk(2)][lane(32)][e(16)]  (4 KB)
// so a lane's v16bf fragment is one contiguous 32-byte LDS read.
//
// Inverse-mapping facts used by the staging code:
//   A (16x32 bf16 frag): tile col kk -> lane_hi = (kk>>3)&1, e = 8*((kk>>4)&1)
//     + (kk&7).  8 consecutive cols in an 8-aligned group => same lane,
//     consecutive e  => one v8bf (b128) LDS store per 8 converted floats.
//   B (32x16 bf16 frag): tile row kk, col c -> lane = ((kk>>3)&1)*16 + (c&15),
//     e = 8*((kk>>4)&1) + (kk&7).  4 consecutive rows in a 4-aligned group => same lane,
//     consecutive e  => one v4bf (b64) LDS store per 4 converted floats.

__global__ __launch_bounds__(256)
void gemm_bf16_wmma(const float* __restrict__ A, const float* __restrict__ W,
                    float* __restrict__ C, const float* __restrict__ bias,
                    int M, int N, int K, int accumulate) {
  __shared__ __bf16 AsF[2 * 4 * 32 * 16];
  __shared__ __bf16 BsF[2 * 2 * 32 * 16];

  const int tid  = threadIdx.x;
  const int lane = tid & 31;
  const int wave = tid >> 5;
  const int mblk = wave & 3;
  const int nblk = wave >> 2;
  const int tileM = blockIdx.y * 64;
  const int tileN = blockIdx.x * 32;

  v8f acc = {0.f, 0.f, 0.f, 0.f, 0.f, 0.f, 0.f, 0.f};

  for (int k0 = 0; k0 < K; k0 += 64) {
    // ---- stage A: 64 rows x 64 K-cols, fp32 -> bf16, fragment layout ----
    // 4096 elements / 256 threads = 2 rounds of 8 consecutive floats.
#pragma unroll
    for (int ro = 0; ro < 2; ++ro) {
      int idx = tid + ro * 256;
      int rr = idx >> 3, oct = idx & 7;     // row in tile, 8-col group
      int gr = tileM + rr;
      v8bf pk;
      if (gr < M) {
        const float4* ap = (const float4*)(A + (size_t)gr * K + k0 + oct * 8);
        float4 x0 = ap[0], x1 = ap[1];      // two coalesced global_load_b128
        pk[0] = (__bf16)x0.x; pk[1] = (__bf16)x0.y;
        pk[2] = (__bf16)x0.z; pk[3] = (__bf16)x0.w;
        pk[4] = (__bf16)x1.x; pk[5] = (__bf16)x1.y;
        pk[6] = (__bf16)x1.z; pk[7] = (__bf16)x1.w;
      } else {
#pragma unroll
        for (int j = 0; j < 8; ++j) pk[j] = (__bf16)0.f;
      }
      int kslab = oct >> 2, quad = oct & 3;
      int l  = ((quad & 1) << 4) | (rr & 15);
      int eb = (quad >> 1) * 8;
      int blk = rr >> 4;
      *(v8bf*)&AsF[((((kslab * 4 + blk) * 32) + l) << 4) + eb] = pk;   // ds_store_b128
    }
    // ---- stage B: 64 K-rows x 32 N-cols, fragment layout ----
    // 2048 elements / 256 threads = 2 rounds of 4 rows per fixed column.
#pragma unroll
    for (int ro = 0; ro < 2; ++ro) {
      int idx = tid + ro * 256;
      int c = idx & 31, kgrp = idx >> 5;    // consecutive tids -> consecutive cols
      int kk = kgrp << 2;
      const float* wp = W + (size_t)(k0 + kk) * N + tileN + c;
      v4bf pk;
#pragma unroll
      for (int j = 0; j < 4; ++j) pk[j] = (__bf16)wp[(size_t)j * N];
      int kslab = kk >> 5, kkl = kk & 31;
      int l  = (((kkl >> 3) & 1) << 4) | (c & 15);
      int eb = (((kkl >> 4) & 1) << 3) | (kkl & 7);
      int nb = c >> 4;
      *(v4bf*)&BsF[((((kslab * 2 + nb) * 32) + l) << 4) + eb] = pk;    // ds_store_b64
    }
    __syncthreads();

    // speculative prefetch of the next K-tile of A (global_prefetch_b8)
    if (k0 + 64 < K && (tid & 7) == 0) {
      int gr = tileM + (tid >> 3) * 2;
      if (gr < M) __builtin_prefetch(A + (size_t)gr * K + k0 + 64, 0, 1);
    }

    // ---- 2 x WMMA: each lane reads its fragments as contiguous b128 pairs
#pragma unroll
    for (int ks = 0; ks < 2; ++ks) {
      v16bf af = *(const v16bf*)&AsF[(((ks * 4 + mblk) * 32) + lane) << 4];
      v16bf bf = *(const v16bf*)&BsF[(((ks * 2 + nblk) * 32) + lane) << 4];
      acc = __builtin_amdgcn_wmma_f32_16x16x32_bf16(
          /*neg_a=*/false, af, /*neg_b=*/false, bf,
          /*c_mod=*/(short)0, acc, /*reuse_a=*/false, /*reuse_b=*/false);
    }
    __syncthreads();
  }

  // epilogue: C/D layout -- lanes 0-15: M=v, lanes 16-31: M=8+v; col = lane&15
  const int hgrp = lane >> 4, r = lane & 15;
#pragma unroll
  for (int v = 0; v < 8; ++v) {
    int row = tileM + mblk * 16 + (hgrp ? v + 8 : v);
    int col = tileN + nblk * 16 + r;
    if (row < M) {
      size_t off = (size_t)row * N + col;
      float x = acc[v];
      if (bias) x += bias[col];
      if (accumulate) x += C[off];
      C[off] = x;
    }
  }
}

// ------------------------- matvec / small helpers --------------------------

// out[i] = dot(X[i,:], v) ; one wave per node, shuffle reduction
__global__ void matvec_rows_kernel(const float* __restrict__ X,
                                   const float* __restrict__ v,
                                   float* __restrict__ out, int N, int D) {
  int node = (int)(((size_t)blockIdx.x * blockDim.x + threadIdx.x) >> 5);
  int lane = threadIdx.x & 31;
  if (node >= N) return;
  float s = 0.f;
  for (int j = lane; j < D; j += 32) s += X[(size_t)node * D + j] * v[j];
  for (int o = 16; o > 0; o >>= 1) s += __shfl_down(s, o, 32);
  if (lane == 0) out[node] = s;
}

// out[i] = dot(W[i,:], a) for small dd (<=512)
__global__ void small_matvec_kernel(const float* __restrict__ W,
                                    const float* __restrict__ a,
                                    float* __restrict__ out, int dd, int dout) {
  int row = blockIdx.x * blockDim.x + threadIdx.x;
  if (row >= dd) return;
  float s = 0.f;
  for (int j = 0; j < dout; ++j) s += W[(size_t)row * dout + j] * a[j];
  out[row] = s;
}

// ------------------------------- GCN ---------------------------------------

__global__ void gcn_deg_kernel(const int* __restrict__ src, const int* __restrict__ dst,
                               float* __restrict__ deg, int E) {
  int e = blockIdx.x * blockDim.x + threadIdx.x;
  if (e >= E) return;
  if (src[e] != dst[e]) atomicAdd(&deg[dst[e]], 1.f);
}

__global__ void gcn_deg_fin_kernel(float* __restrict__ deg, float* __restrict__ dinv, int N) {
  int i = blockIdx.x * blockDim.x + threadIdx.x;
  if (i >= N) return;
  float d = deg[i] + 1.f;   // re-added self loop
  deg[i] = d;
  dinv[i] = rsqrtf(d);
}

__global__ void gcn_scatter_kernel(const int* __restrict__ src, const int* __restrict__ dst,
                                   const float* __restrict__ dinv, const float* __restrict__ H,
                                   float* __restrict__ G, int E, int D) {
  int e = (int)(((size_t)blockIdx.x * blockDim.x + threadIdx.x) >> 5);
  int lane = threadIdx.x & 31;
  if (e >= E) return;
  int s = src[e], d = dst[e];
  if (s == d) return;
  float c = dinv[s] * dinv[d];
  const float* hp = H + (size_t)s * D;
  float* gp = G + (size_t)d * D;
  for (int f = lane; f < D; f += 32) atomicAdd(&gp[f], c * hp[f]);
}

__global__ void gcn_self_kernel(const float* __restrict__ H, const float* __restrict__ deg,
                                const float* __restrict__ b, float* __restrict__ G,
                                int N, int D) {
  size_t n = (size_t)N * D;
  size_t i = (size_t)blockIdx.x * blockDim.x + threadIdx.x;
  size_t stride = (size_t)gridDim.x * blockDim.x;
  for (; i < n; i += stride) {
    int row = (int)(i / D), col = (int)(i & (D - 1));
    G[i] += H[i] * (1.f / deg[row]) + b[col];
  }
}

// ------------------------------- GAT ---------------------------------------

__device__ __forceinline__ void gat_edge(int idx, const int* src, const int* dst,
                                         int E, int& sa, int& da, bool& isE) {
  if (idx < E) { sa = src[idx]; da = dst[idx]; isE = true; }
  else         { sa = da = idx - E;            isE = false; }
}

__global__ void gat_alpha_kernel(const int* __restrict__ src, const int* __restrict__ dst,
                                 const float* __restrict__ a_s, const float* __restrict__ a_d,
                                 float* __restrict__ alpha, float* __restrict__ mx,
                                 int E, int L) {
  int idx = blockIdx.x * blockDim.x + threadIdx.x;
  if (idx >= E + L) return;
  int sa, da; bool isE;
  gat_edge(idx, src, dst, E, sa, da, isE);
  float a = a_s[sa] + a_d[da];
  a = (a > 0.f) ? a : 0.2f * a;                 // leaky_relu(0.2)
  if (isE && sa == da) a = -1e30f;              // original self-loops masked
  alpha[idx] = a;
  atomicMaxF(&mx[da], a);
}

__global__ void gat_exp_kernel(const int* __restrict__ src, const int* __restrict__ dst,
                               float* __restrict__ alpha, const float* __restrict__ mx,
                               float* __restrict__ ssum, int E, int L) {
  int idx = blockIdx.x * blockDim.x + threadIdx.x;
  if (idx >= E + L) return;
  int sa, da; bool isE;
  gat_edge(idx, src, dst, E, sa, da, isE);
  float ex = expf(alpha[idx] - mx[da]);
  alpha[idx] = ex;
  atomicAdd(&ssum[da], ex);
}

__global__ void gat_scatter_kernel(const int* __restrict__ src, const int* __restrict__ dst,
                                   const float* __restrict__ alpha, const float* __restrict__ ssum,
                                   const float* __restrict__ Hs, float* __restrict__ G,
                                   int E, int L, int D) {
  int idx = (int)(((size_t)blockIdx.x * blockDim.x + threadIdx.x) >> 5);
  int lane = threadIdx.x & 31;
  if (idx >= E + L) return;
  int sa, da; bool isE;
  gat_edge(idx, src, dst, E, sa, da, isE);
  float w = alpha[idx] / (ssum[da] + 1e-16f);
  const float* hp = Hs + (size_t)sa * D;
  float* gp = G + (size_t)da * D;
  for (int f = lane; f < D; f += 32) atomicAdd(&gp[f], w * hp[f]);
}

// ------------------------------- SAGE --------------------------------------

__global__ void sage_scatter_kernel(const int* __restrict__ src, const int* __restrict__ dst,
                                    const float* __restrict__ X, float* __restrict__ S,
                                    float* __restrict__ cnt, int E, int Din) {
  int e = (int)(((size_t)blockIdx.x * blockDim.x + threadIdx.x) >> 5);
  int lane = threadIdx.x & 31;
  if (e >= E) return;
  int s = src[e], d = dst[e];
  if (lane == 0) atomicAdd(&cnt[d], 1.f);
  const float* xp = X + (size_t)s * Din;
  float* sp = S + (size_t)d * Din;
  for (int f = lane; f < Din; f += 32) atomicAdd(&sp[f], xp[f]);
}

__global__ void sage_mean_kernel(float* __restrict__ S, const float* __restrict__ cnt,
                                 int N, int Din) {
  size_t n = (size_t)N * Din;
  size_t i = (size_t)blockIdx.x * blockDim.x + threadIdx.x;
  size_t stride = (size_t)gridDim.x * blockDim.x;
  for (; i < n; i += stride) {
    int row = (int)(i / Din);
    S[i] /= fmaxf(cnt[row], 1.f);
  }
}

// ----------------------------- host side -----------------------------------

namespace {

constexpr int NG = 20000, NM = 8000, NR = 2000, NB = 12000;
constexpr int FG = 512, FO = 256, HID = 256, OUT = 128;

struct GatP  { const float *Wd, *Ws, *ad, *as_, *b; };
struct SageP { const float *Wl, *Wr, *bl; };
struct GcnP  { const float *W, *b; };

struct Scratch {
  float *Xg1, *Xm1, *Xr1, *Xb1;     // layer-0 outputs / layer-1 inputs
  float *Hbuf;                       // per-conv h = x@W (max 20000x256)
  float *SageBuf;                    // neighbor-sum buffer (max 12000x512)
  float *deg, *dinv, *cnt;
  float *wvec, *a_s, *a_d, *mx, *ssum, *alpha;
};

inline void fill(hipStream_t st, float* p, float v, size_t n) {
  int blocks = (int)((n + 255) / 256); if (blocks > 4096) blocks = 4096;
  fill_kernel<<<blocks, 256, 0, st>>>(p, v, n);
}

inline void gemm(hipStream_t st, const float* A, const float* W, float* C,
                 int M, int N, int K, const float* bias, int accum) {
  dim3 g(cdiv_h(N, 32), cdiv_h(M, 64));
  gemm_bf16_wmma<<<g, 256, 0, st>>>(A, W, C, bias, M, N, K, accum);
}

void run_gcn(hipStream_t st, Scratch& ws, const float* x, int N, int din,
             const GcnP& p, const int* ei, int E, int dout, float* G) {
  gemm(st, x, p.W, ws.Hbuf, N, dout, din, nullptr, 0);
  const int *src = ei, *dst = ei + E;
  fill(st, ws.deg, 0.f, N);
  gcn_deg_kernel<<<cdiv_h(E, 256), 256, 0, st>>>(src, dst, ws.deg, E);
  gcn_deg_fin_kernel<<<cdiv_h(N, 256), 256, 0, st>>>(ws.deg, ws.dinv, N);
  gcn_scatter_kernel<<<cdiv_h(E, 8), 256, 0, st>>>(src, dst, ws.dinv, ws.Hbuf, G, E, dout);
  int blocks = cdiv_h(N * dout, 256); if (blocks > 4096) blocks = 4096;
  gcn_self_kernel<<<blocks, 256, 0, st>>>(ws.Hbuf, ws.deg, p.b, G, N, dout);
}

void run_gat(hipStream_t st, Scratch& ws,
             const float* xs, int ns, int din_s,
             const float* xd, int nd, int din_d,
             const GatP& p, const int* ei, int E, int dout, float* G) {
  const int L = (ns < nd) ? ns : nd;
  const int *src = ei, *dst = ei + E;
  // hs = xs @ Ws (also the message features)
  gemm(st, xs, p.Ws, ws.Hbuf, ns, dout, din_s, nullptr, 0);
  // a_d = xd @ (Wd @ ad)   -- hd never materialized
  small_matvec_kernel<<<cdiv_h(din_d, 128), 128, 0, st>>>(p.Wd, p.ad, ws.wvec, din_d, dout);
  matvec_rows_kernel<<<cdiv_h(nd, 8), 256, 0, st>>>(xd, ws.wvec, ws.a_d, nd, din_d);
  // a_s = hs @ as
  matvec_rows_kernel<<<cdiv_h(ns, 8), 256, 0, st>>>(ws.Hbuf, p.as_, ws.a_s, ns, dout);
  // segment softmax over dst
  fill(st, ws.mx, -1e38f, nd);
  gat_alpha_kernel<<<cdiv_h(E + L, 256), 256, 0, st>>>(src, dst, ws.a_s, ws.a_d,
                                                       ws.alpha, ws.mx, E, L);
  fill(st, ws.ssum, 0.f, nd);
  gat_exp_kernel<<<cdiv_h(E + L, 256), 256, 0, st>>>(src, dst, ws.alpha, ws.mx, ws.ssum, E, L);
  gat_scatter_kernel<<<cdiv_h(E + L, 8), 256, 0, st>>>(src, dst, ws.alpha, ws.ssum,
                                                       ws.Hbuf, G, E, L, dout);
  int blocks = cdiv_h(nd * dout, 256); if (blocks > 4096) blocks = 4096;
  bias_add_kernel<<<blocks, 256, 0, st>>>(G, p.b, nd, dout);
}

void run_sage(hipStream_t st, Scratch& ws,
              const float* xs, int din_s,
              const float* xd, int nd, int din_d,
              const SageP& p, const int* ei, int E, int dout, float* G) {
  const int *src = ei, *dst = ei + E;
  fill(st, ws.SageBuf, 0.f, (size_t)nd * din_s);
  fill(st, ws.cnt, 0.f, nd);
  sage_scatter_kernel<<<cdiv_h(E, 8), 256, 0, st>>>(src, dst, xs, ws.SageBuf, ws.cnt, E, din_s);
  int blocks = cdiv_h(nd * din_s, 256); if (blocks > 4096) blocks = 4096;
  sage_mean_kernel<<<blocks, 256, 0, st>>>(ws.SageBuf, ws.cnt, nd, din_s);
  gemm(st, ws.SageBuf, p.Wl, G, nd, dout, din_s, p.bl, 1);   // G += mean@Wl + bl
  gemm(st, xd, p.Wr, G, nd, dout, din_d, nullptr, 1);        // G += xd@Wr
}

} // namespace

extern "C" void kernel_launch(void* const* d_in, const int* in_sizes, int n_in,
                              void* d_out, int out_size, void* d_ws, size_t ws_size,
                              hipStream_t stream) {
  (void)n_in; (void)out_size; (void)ws_size;

  // ---- inputs (setup_inputs dict order) ----
  const float* x_gene = (const float*)d_in[0];
  const float* x_msig = (const float*)d_in[1];
  const float* x_react = (const float*)d_in[2];
  const float* x_bp = (const float*)d_in[3];
  const int* ei_g2g = (const int*)d_in[4];   int E_g2g = in_sizes[4] / 2;
  const int* ei_gm  = (const int*)d_in[5];   int E_gm  = in_sizes[5] / 2;
  const int* ei_gr  = (const int*)d_in[6];   int E_gr  = in_sizes[6] / 2;
  const int* ei_gb  = (const int*)d_in[7];   int E_gb  = in_sizes[7] / 2;
  const int* ei_mg  = (const int*)d_in[8];   int E_mg  = in_sizes[8] / 2;
  const int* ei_rg  = (const int*)d_in[9];   int E_rg  = in_sizes[9] / 2;
  const int* ei_bg  = (const int*)d_in[10];  int E_bg  = in_sizes[10] / 2;

  // ---- params: JAX pytree flattening sorts dict keys alphabetically ----
  // per layer: g2g{W,b}, genebp{Wl,Wr,bl}, genemsig{Wd,Ws,ad,as,b},
  //            genereact{...}, rev_genebp{Wl,Wr,bl}, rev_genemsig{...},
  //            rev_genereact{...}
  int pi = 11;
  auto nxt = [&]() { return (const float*)d_in[pi++]; };
  GcnP g2g[2]; SageP genebp[2], rev_genebp[2];
  GatP genemsig[2], genereact[2], rev_genemsig[2], rev_genereact[2];
  for (int l = 0; l < 2; ++l) {
    g2g[l].W = nxt(); g2g[l].b = nxt();
    genebp[l].Wl = nxt(); genebp[l].Wr = nxt(); genebp[l].bl = nxt();
    genemsig[l].Wd = nxt(); genemsig[l].Ws = nxt();
    genemsig[l].ad = nxt(); genemsig[l].as_ = nxt(); genemsig[l].b = nxt();
    genereact[l].Wd = nxt(); genereact[l].Ws = nxt();
    genereact[l].ad = nxt(); genereact[l].as_ = nxt(); genereact[l].b = nxt();
    rev_genebp[l].Wl = nxt(); rev_genebp[l].Wr = nxt(); rev_genebp[l].bl = nxt();
    rev_genemsig[l].Wd = nxt(); rev_genemsig[l].Ws = nxt();
    rev_genemsig[l].ad = nxt(); rev_genemsig[l].as_ = nxt(); rev_genemsig[l].b = nxt();
    rev_genereact[l].Wd = nxt(); rev_genereact[l].Ws = nxt();
    rev_genereact[l].ad = nxt(); rev_genereact[l].as_ = nxt(); rev_genereact[l].b = nxt();
  }

  // ---- workspace bump allocation ----
  char* wp = (char*)d_ws;
  auto alloc = [&](size_t nfloats) {
    float* p = (float*)wp; wp += nfloats * sizeof(float); return p;
  };
  Scratch ws;
  ws.Xg1 = alloc((size_t)NG * HID);
  ws.Xm1 = alloc((size_t)NM * HID);
  ws.Xr1 = alloc((size_t)NR * HID);
  ws.Xb1 = alloc((size_t)NB * HID);
  ws.Hbuf = alloc((size_t)NG * HID);
  ws.SageBuf = alloc((size_t)NB * FG > (size_t)NG * FO ? (size_t)NB * FG : (size_t)NG * FO);
  ws.deg = alloc(NG); ws.dinv = alloc(NG); ws.cnt = alloc(NG);
  ws.wvec = alloc(FG);
  ws.a_s = alloc(NG); ws.a_d = alloc(NG);
  ws.mx = alloc(NG); ws.ssum = alloc(NG);
  ws.alpha = alloc(680000);   // max E + L = 400000 + 8000

  float* out = (float*)d_out;
  float* out_g = out;
  float* out_m = out_g + (size_t)NG * OUT;
  float* out_r = out_m + (size_t)NM * OUT;
  float* out_b = out_r + (size_t)NR * OUT;

  for (int l = 0; l < 2; ++l) {
    const int din_g = l ? HID : FG;
    const int din_o = l ? HID : FO;
    const int dout  = l ? OUT : HID;
    const float* xg = l ? ws.Xg1 : x_gene;
    const float* xm = l ? ws.Xm1 : x_msig;
    const float* xr = l ? ws.Xr1 : x_react;
    const float* xb = l ? ws.Xb1 : x_bp;
    float* Gg = l ? out_g : ws.Xg1;
    float* Gm = l ? out_m : ws.Xm1;
    float* Gr = l ? out_r : ws.Xr1;
    float* Gb = l ? out_b : ws.Xb1;

    fill(stream, Gg, 0.f, (size_t)NG * dout);
    fill(stream, Gm, 0.f, (size_t)NM * dout);
    fill(stream, Gr, 0.f, (size_t)NR * dout);
    fill(stream, Gb, 0.f, (size_t)NB * dout);

    // gene = GCN(g2g) + GAT(msig->gene) + GAT(react->gene) + SAGE(bp->gene)
    run_gcn(stream, ws, xg, NG, din_g, g2g[l], ei_g2g, E_g2g, dout, Gg);
    run_gat(stream, ws, xm, NM, din_o, xg, NG, din_g, rev_genemsig[l], ei_mg, E_mg, dout, Gg);
    run_gat(stream, ws, xr, NR, din_o, xg, NG, din_g, rev_genereact[l], ei_rg, E_rg, dout, Gg);
    run_sage(stream, ws, xb, din_o, xg, NG, din_g, rev_genebp[l], ei_bg, E_bg, dout, Gg);
    // msig / reactome / bp
    run_gat(stream, ws, xg, NG, din_g, xm, NM, din_o, genemsig[l], ei_gm, E_gm, dout, Gm);
    run_gat(stream, ws, xg, NG, din_g, xr, NR, din_o, genereact[l], ei_gr, E_gr, dout, Gr);
    run_sage(stream, ws, xg, din_g, xb, NB, din_o, genebp[l], ei_gb, E_gb, dout, Gb);

    relu_kernel<<<4096, 256, 0, stream>>>(Gg, (size_t)NG * dout);
    relu_kernel<<<4096, 256, 0, stream>>>(Gm, (size_t)NM * dout);
    relu_kernel<<<4096, 256, 0, stream>>>(Gr, (size_t)NR * dout);
    relu_kernel<<<4096, 256, 0, stream>>>(Gb, (size_t)NB * dout);
  }
}
